// EncodeProcessDecode_23038204576041
// MI455X (gfx1250) — matrix operand order repaired and checked by reading, not compiled, not run on hardware
//
#include <hip/hip_runtime.h>

// ---------------------------------------------------------------------------
// EncodeProcessDecode GNN on MI455X (gfx1250).
// GEMMs on v_wmma_f32_16x16x32_bf16 (bf16 in, fp32 accumulate).
// B-weight tiles streamed with global_load_async_to_lds_b128 (ASYNCcnt),
// double-buffered in LDS and overlapped with WMMA.
// ~184 GFLOP; all state L2-resident -> matrix-core bound.
// ---------------------------------------------------------------------------

typedef __bf16        v16bf __attribute__((ext_vector_type(16)));
typedef float         v8f   __attribute__((ext_vector_type(8)));
typedef unsigned int  u32x4 __attribute__((ext_vector_type(4)));

#define NODE_F 30
#define EDGE_F 4
#define LDIM   128
#define OUTF   3
#define STEPS  10

// Low 32 bits of a flat pointer to LDS == LDS byte offset (ISA 10.2 aperture).
__device__ __forceinline__ unsigned ldsOffsetOf(const void* p) {
  return (unsigned)(unsigned long long)p;
}

// Async copy 16B global -> LDS per active lane (tracked by ASYNCcnt).
__device__ __forceinline__ void asyncCopyB128(unsigned ldsByteOff, const void* gbase,
                                              int gByteOff) {
  asm volatile("global_load_async_to_lds_b128 %0, %1, %2"
               :: "v"(ldsByteOff), "v"(gByteOff), "s"(gbase)
               : "memory");
}

template <int N>
__device__ __forceinline__ void waitAsyncLE() {
  asm volatile("s_wait_asynccnt %0" :: "n"(N) : "memory");
}

// Issue async loads for one packed-B K-chunk (ntiles * 1KB) into bst[buf].
__device__ __forceinline__ void issueB(const __bf16* Wp, int kc, int buf, int ntiles,
                                       unsigned bstOff, int tid) {
  const char* g = (const char*)(Wp + (size_t)kc * ((size_t)ntiles * 512));
  for (int i = tid; i < ntiles * 64; i += 256)
    asyncCopyB128(bstOff + (unsigned)buf * 8192u + (unsigned)i * 16u, g, i * 16);
}

// ---------------------------------------------------------------------------
// Weight prep: fp32 [K][N] -> bf16 packed in per-lane B-fragment order.
// B 32x16 tile (contiguous k order chosen here, matched by A): lanes 0-15 hold K=0..15 (N=lane), lanes 16-31
// hold K=16..31; within a lane, slot s <-> k%16.
// packed[( (kc*ntiles + nt)*32 + lane )*16 + slot]
// ---------------------------------------------------------------------------
__global__ void pack_w_k(const float* __restrict__ src, __bf16* __restrict__ dst,
                         int K, int N, int KPAD, int NPAD) {
  int p = blockIdx.x * 256 + threadIdx.x;
  int total = KPAD * NPAD;
  if (p >= total) return;
  int slot = p & 15;
  int lane = (p >> 4) & 31;
  int blk  = p >> 9;
  int ntiles = NPAD >> 4;
  int kc = blk / ntiles;
  int nt = blk - kc * ntiles;
  int k = kc * 32 + ((lane >> 4) << 4) + slot;
  int n = nt * 16 + (lane & 15);
  float v = (k < K && n < N) ? src[k * N + n] : 0.f;
  dst[p] = (__bf16)v;
}

__global__ void zero_k(float* p, int n) {
  int i = blockIdx.x * 256 + threadIdx.x;
  if (i < n) p[i] = 0.f;
}

// ---------------------------------------------------------------------------
// Layer-0 scalar fetch (modes with ragged K only).
// ---------------------------------------------------------------------------
template <int MODE>
__device__ __forceinline__ float fetch0(const float* in0, int gRow, int kG) {
  if constexpr (MODE == 0) return (kG < NODE_F) ? in0[gRow * NODE_F + kG] : 0.f;
  else                     return (kG < EDGE_F) ? in0[gRow * EDGE_F + kG] : 0.f;
}

// ---------------------------------------------------------------------------
// Fused 3-layer MLP (+optional LayerNorm + mode epilogue) over a 128-row block.
// 8 wave32s; wave w owns output rows [16w,16w+16), all 128 cols.
// MODE: 0 enc-node -> hx ; 1 enc-edge -> he ; 2 proc-edge (he+=ue, agg atomic)
//       3 proc-node (hx+=un) ; 4 decoder -> out[N,3]
// ---------------------------------------------------------------------------
template <int MODE, int K0PAD, int NT2, bool DO_LN>
__global__ __launch_bounds__(256) void mlp_k(
    const float* __restrict__ in0, const float* in1,
    const int* __restrict__ eSrc, const int* __restrict__ eDst,
    const __bf16* __restrict__ Wp0, const __bf16* __restrict__ Wp1,
    const __bf16* __restrict__ Wp2,
    const float* __restrict__ b0, const float* __restrict__ b1,
    const float* __restrict__ b2,
    const float* __restrict__ g, const float* __restrict__ bt,
    float* out, float* agg, int nrows)
{
  __shared__ __attribute__((aligned(16))) __bf16 hid[128 * 136];   // 34.8 KB
  __shared__ __attribute__((aligned(16))) __bf16 ast[128 * 32];    // 8 KB
  __shared__ __attribute__((aligned(16))) __bf16 bst[2][8 * 512];  // 16 KB
  __shared__ int sIdx[2][128];

  const int tid  = threadIdx.x;
  const int wv   = tid >> 5;
  const int lane = tid & 31;
  const int h    = lane >> 4;
  const int ln16 = lane & 15;
  const int row0 = blockIdx.x * 128;
  const unsigned bstOff = ldsOffsetOf(&bst[0][0]);

  if constexpr (MODE == 2) {
    if (tid < 128) {
      int e = row0 + tid;
      sIdx[0][tid] = (e < nrows) ? eSrc[e] : 0;
      sIdx[1][tid] = (e < nrows) ? eDst[e] : 0;
    }
  }
  __syncthreads();

  v8f acc[8];

  // ============================ layer 0 =================================
#pragma unroll
  for (int t = 0; t < 8; ++t)
#pragma unroll
    for (int i = 0; i < 8; ++i) acc[t][i] = 0.f;

  constexpr int KC0 = K0PAD / 32;
  issueB(Wp0, 0, 0, 8, bstOff, tid);
  for (int kc = 0; kc < KC0; ++kc) {
    { // stage A chunk [128 rows x 32 K] as bf16 (gather + pad + convert)
      int r  = tid >> 1;
      int kb = (tid & 1) << 4;
      int gRow = row0 + r;
      bool ok = gRow < nrows;
      float v[16];
      if constexpr (MODE == 0 || MODE == 1) {
#pragma unroll
        for (int i = 0; i < 16; ++i)
          v[i] = ok ? fetch0<MODE>(in0, gRow, kc * 32 + kb + i) : 0.f;
      } else {
        int kG0 = kc * 32 + kb;   // 16-float run never crosses a 128-segment
        const float* base;
        if constexpr (MODE == 2) {
          if (kG0 < LDIM)          base = in0 + (size_t)sIdx[0][r] * LDIM + kG0;
          else if (kG0 < 2 * LDIM) base = in0 + (size_t)sIdx[1][r] * LDIM + (kG0 - LDIM);
          else                     base = in1 + (size_t)gRow * LDIM + (kG0 - 2 * LDIM);
        } else if constexpr (MODE == 3) {
          base = (kG0 < LDIM) ? in0 + (size_t)gRow * LDIM + kG0
                              : in1 + (size_t)gRow * LDIM + (kG0 - LDIM);
        } else {
          base = in0 + (size_t)gRow * LDIM + kG0;
        }
        if (ok) {
#pragma unroll
          for (int q = 0; q < 4; ++q) {
            float4 f = *(const float4*)(base + q * 4);
            v[q * 4 + 0] = f.x; v[q * 4 + 1] = f.y;
            v[q * 4 + 2] = f.z; v[q * 4 + 3] = f.w;
          }
        } else {
#pragma unroll
          for (int i = 0; i < 16; ++i) v[i] = 0.f;
        }
      }
      unsigned up[8];
#pragma unroll
      for (int j = 0; j < 8; ++j) {
        unsigned short l0 = __builtin_bit_cast(unsigned short, (__bf16)v[2 * j]);
        unsigned short l1 = __builtin_bit_cast(unsigned short, (__bf16)v[2 * j + 1]);
        up[j] = (unsigned)l0 | ((unsigned)l1 << 16);
      }
      u32x4* dp = (u32x4*)(ast + r * 32 + kb);
      dp[0] = u32x4{up[0], up[1], up[2], up[3]};
      dp[1] = u32x4{up[4], up[5], up[6], up[7]};
    }
    if (kc + 1 < KC0) { issueB(Wp0, kc + 1, (kc + 1) & 1, 8, bstOff, tid); waitAsyncLE<2>(); }
    else              { waitAsyncLE<0>(); }
    __syncthreads();

    union { v16bf v; u32x4 u[2]; } a;
    { // A fragment: 2x ds_load_b128, lane half h holds K=16h..16h+15 (matches packed B)
      const __bf16* ab = ast + (wv * 16 + ln16) * 32 + h * 16;
      a.u[0] = *(const u32x4*)(ab);
      a.u[1] = *(const u32x4*)(ab + 8);
    }
#pragma unroll
    for (int t = 0; t < 8; ++t) {
      union { v16bf v; u32x4 u[2]; } b;
      const __bf16* bb = &bst[kc & 1][(t * 32 + lane) * 16];
      b.u[0] = *(const u32x4*)(bb);
      b.u[1] = *(const u32x4*)(bb + 8);
      acc[t] = __builtin_amdgcn_wmma_f32_16x16x32_bf16(
          false, a.v, false, b.v, (short)0, acc[t], false, false);
    }
    __syncthreads();
  }
  // bias + ReLU -> hid (bf16). Wave w touches only its own rows.
#pragma unroll
  for (int t = 0; t < 8; ++t) {
    float bv = b0[t * 16 + ln16];
#pragma unroll
    for (int r = 0; r < 8; ++r) {
      float v = fmaxf(acc[t][r] + bv, 0.f);
      hid[(wv * 16 + h * 8 + r) * 136 + t * 16 + ln16] = (__bf16)v;
    }
  }
  __syncthreads();

  // ====================== layers 1 and 2 ================================
#pragma unroll
  for (int layer = 1; layer <= 2; ++layer) {
    const __bf16* Wp = (layer == 1) ? Wp1 : Wp2;
    const float* bb_ = (layer == 1) ? b1 : b2;
    const int NT = (layer == 1) ? 8 : NT2;
#pragma unroll
    for (int t = 0; t < 8; ++t)
#pragma unroll
      for (int i = 0; i < 8; ++i) acc[t][i] = 0.f;

    issueB(Wp, 0, 0, NT, bstOff, tid);
    for (int kc = 0; kc < 4; ++kc) {
      if (kc + 1 < 4) {
        issueB(Wp, kc + 1, (kc + 1) & 1, NT, bstOff, tid);
        if (NT == 8) waitAsyncLE<2>(); else waitAsyncLE<1>();
      } else {
        waitAsyncLE<0>();
      }
      __syncthreads();
      union { v16bf v; u32x4 u[2]; } a;
      {
        const __bf16* ab = hid + (wv * 16 + ln16) * 136 + kc * 32 + h * 16;
        a.u[0] = *(const u32x4*)(ab);
        a.u[1] = *(const u32x4*)(ab + 8);
      }
#pragma unroll
      for (int t = 0; t < 8; ++t) {
        if (t >= NT) break;
        union { v16bf v; u32x4 u[2]; } b;
        const __bf16* bb = &bst[kc & 1][(t * 32 + lane) * 16];
        b.u[0] = *(const u32x4*)(bb);
        b.u[1] = *(const u32x4*)(bb + 8);
        acc[t] = __builtin_amdgcn_wmma_f32_16x16x32_bf16(
            false, a.v, false, b.v, (short)0, acc[t], false, false);
      }
      __syncthreads();
    }
    if (layer == 1) { // bias + ReLU -> hid
#pragma unroll
      for (int t = 0; t < 8; ++t) {
        float bv = bb_[t * 16 + ln16];
#pragma unroll
        for (int r = 0; r < 8; ++r) {
          float v = fmaxf(acc[t][r] + bv, 0.f);
          hid[(wv * 16 + h * 8 + r) * 136 + t * 16 + ln16] = (__bf16)v;
        }
      }
      __syncthreads();
    }
  }

  // ============================ epilogue ================================
  if constexpr (!DO_LN) {
    // decoder: single padded 16-col tile, cols 0..2 valid
#pragma unroll
    for (int r = 0; r < 8; ++r) {
      int gRow = row0 + wv * 16 + h * 8 + r;
      if (gRow < nrows && ln16 < OUTF)
        out[gRow * OUTF + ln16] = acc[0][r] + b2[ln16];
    }
  } else {
#pragma unroll
    for (int t = 0; t < 8; ++t) {
      float bv = b2[t * 16 + ln16];
#pragma unroll
      for (int r = 0; r < 8; ++r) acc[t][r] += bv;
    }
#pragma unroll
    for (int r = 0; r < 8; ++r) {
      float s = 0.f, s2 = 0.f;
#pragma unroll
      for (int t = 0; t < 8; ++t) { float v = acc[t][r]; s += v; s2 += v * v; }
#pragma unroll
      for (int m = 1; m < 16; m <<= 1) {  // reduce within 16-lane half
        s  += __shfl_xor(s,  m, 32);
        s2 += __shfl_xor(s2, m, 32);
      }
      float mean = s * (1.f / 128.f);
      float var  = s2 * (1.f / 128.f) - mean * mean;
      float inv  = rsqrtf(var + 1e-5f);
      int gRow = row0 + wv * 16 + h * 8 + r;
      bool valid = gRow < nrows;
#pragma unroll
      for (int t = 0; t < 8; ++t) {
        int col = t * 16 + ln16;
        float val = (acc[t][r] - mean) * inv * g[col] + bt[col];
        if (valid) {
          if constexpr (MODE == 0 || MODE == 1) {
            out[gRow * LDIM + col] = val;
          } else if constexpr (MODE == 2) {
            out[gRow * LDIM + col] = in1[gRow * LDIM + col] + val;  // he += ue
            atomicAdd(&agg[(size_t)sIdx[1][wv * 16 + h * 8 + r] * LDIM + col], val);
          } else if constexpr (MODE == 3) {
            out[gRow * LDIM + col] += val;                          // hx += un
          }
        }
      }
    }
  }
}

// ---------------------------------------------------------------------------
extern "C" void kernel_launch(void* const* d_in, const int* in_sizes, int n_in,
                              void* d_out, int out_size, void* d_ws, size_t ws_size,
                              hipStream_t stream) {
  const float* x   = (const float*)d_in[0];
  const float* ea  = (const float*)d_in[1];
  const int*   ei  = (const int*)d_in[2];
  const float* enW0=(const float*)d_in[3];  const float* enb0=(const float*)d_in[4];
  const float* enW1=(const float*)d_in[5];  const float* enb1=(const float*)d_in[6];
  const float* enW2=(const float*)d_in[7];  const float* enb2=(const float*)d_in[8];
  const float* eng =(const float*)d_in[9];  const float* enbt=(const float*)d_in[10];
  const float* eeW0=(const float*)d_in[11]; const float* eeb0=(const float*)d_in[12];
  const float* eeW1=(const float*)d_in[13]; const float* eeb1=(const float*)d_in[14];
  const float* eeW2=(const float*)d_in[15]; const float* eeb2=(const float*)d_in[16];
  const float* eeg =(const float*)d_in[17]; const float* eebt=(const float*)d_in[18];
  const float* peW0=(const float*)d_in[19]; const float* peb0=(const float*)d_in[20];
  const float* peW1=(const float*)d_in[21]; const float* peb1=(const float*)d_in[22];
  const float* peW2=(const float*)d_in[23]; const float* peb2=(const float*)d_in[24];
  const float* peg =(const float*)d_in[25]; const float* pebt=(const float*)d_in[26];
  const float* pnW0=(const float*)d_in[27]; const float* pnb0=(const float*)d_in[28];
  const float* pnW1=(const float*)d_in[29]; const float* pnb1=(const float*)d_in[30];
  const float* pnW2=(const float*)d_in[31]; const float* pnb2=(const float*)d_in[32];
  const float* png =(const float*)d_in[33]; const float* pnbt=(const float*)d_in[34];
  const float* dW0 =(const float*)d_in[35]; const float* db0 =(const float*)d_in[36];
  const float* dW1 =(const float*)d_in[37]; const float* db1 =(const float*)d_in[38];
  const float* dW2 =(const float*)d_in[39]; const float* db2 =(const float*)d_in[40];

  const int N = in_sizes[0] / NODE_F;
  const int E = in_sizes[1] / EDGE_F;
  const int* eSrc = ei;
  const int* eDst = ei + E;

  // workspace layout: fp32 state buffers, then packed bf16 weights (~64.7 MB)
  float* hx  = (float*)d_ws;
  float* he  = hx + (size_t)N * LDIM;
  float* agg = he + (size_t)E * LDIM;
  __bf16* wcur = (__bf16*)(agg + (size_t)N * LDIM);

  auto pack = [&](const float* s, int K, int Nc, int KPAD, int NPAD) -> __bf16* {
    __bf16* d = wcur;
    int total = KPAD * NPAD;
    pack_w_k<<<(total + 255) / 256, 256, 0, stream>>>(s, d, K, Nc, KPAD, NPAD);
    wcur += total;
    return d;
  };

  __bf16* enW0p = pack(enW0, NODE_F, LDIM, 32, 128);
  __bf16* enW1p = pack(enW1, 128, 128, 128, 128);
  __bf16* enW2p = pack(enW2, 128, 128, 128, 128);
  __bf16* eeW0p = pack(eeW0, EDGE_F, LDIM, 32, 128);
  __bf16* eeW1p = pack(eeW1, 128, 128, 128, 128);
  __bf16* eeW2p = pack(eeW2, 128, 128, 128, 128);
  __bf16 *peW0p[STEPS], *peW1p[STEPS], *peW2p[STEPS];
  __bf16 *pnW0p[STEPS], *pnW1p[STEPS], *pnW2p[STEPS];
  for (int s = 0; s < STEPS; ++s) {
    peW0p[s] = pack(peW0 + (size_t)s * 384 * 128, 384, 128, 384, 128);
    peW1p[s] = pack(peW1 + (size_t)s * 128 * 128, 128, 128, 128, 128);
    peW2p[s] = pack(peW2 + (size_t)s * 128 * 128, 128, 128, 128, 128);
    pnW0p[s] = pack(pnW0 + (size_t)s * 256 * 128, 256, 128, 256, 128);
    pnW1p[s] = pack(pnW1 + (size_t)s * 128 * 128, 128, 128, 128, 128);
    pnW2p[s] = pack(pnW2 + (size_t)s * 128 * 128, 128, 128, 128, 128);
  }
  __bf16* dW0p = pack(dW0, 128, 128, 128, 128);
  __bf16* dW1p = pack(dW1, 128, 128, 128, 128);
  __bf16* dW2p = pack(dW2, 128, OUTF, 128, 16);

  const int nbN = (N + 127) / 128;
  const int nbE = (E + 127) / 128;

  // encoders
  mlp_k<0, 32, 8, true><<<nbN, 256, 0, stream>>>(
      x, nullptr, nullptr, nullptr, enW0p, enW1p, enW2p,
      enb0, enb1, enb2, eng, enbt, hx, nullptr, N);
  mlp_k<1, 32, 8, true><<<nbE, 256, 0, stream>>>(
      ea, nullptr, nullptr, nullptr, eeW0p, eeW1p, eeW2p,
      eeb0, eeb1, eeb2, eeg, eebt, he, nullptr, E);

  // processor: S interaction steps
  for (int s = 0; s < STEPS; ++s) {
    zero_k<<<(N * LDIM + 255) / 256, 256, 0, stream>>>(agg, N * LDIM);
    mlp_k<2, 384, 8, true><<<nbE, 256, 0, stream>>>(
        hx, he, eSrc, eDst, peW0p[s], peW1p[s], peW2p[s],
        peb0 + s * 128, peb1 + s * 128, peb2 + s * 128,
        peg + s * 128, pebt + s * 128, he, agg, E);
    mlp_k<3, 256, 8, true><<<nbN, 256, 0, stream>>>(
        hx, agg, nullptr, nullptr, pnW0p[s], pnW1p[s], pnW2p[s],
        pnb0 + s * 128, pnb1 + s * 128, pnb2 + s * 128,
        png + s * 128, pnbt + s * 128, hx, nullptr, N);
  }

  // decoder
  mlp_k<4, 128, 1, false><<<nbN, 256, 0, stream>>>(
      hx, nullptr, nullptr, nullptr, dW0p, dW1p, dW2p,
      db0, db1, db2, nullptr, nullptr, (float*)d_out, nullptr, N);

  (void)n_in; (void)out_size; (void)ws_size;
}
